// CausalSelfAttention_31593779429522
// MI455X (gfx1250) — matrix operand order repaired and checked by reading, not compiled, not run on hardware
//
#include <hip/hip_runtime.h>
#include <hip/hip_bf16.h>

#define L_SEQ 2048
#define DMODEL 2048
#define NHEAD 32
#define NKV 8
#define HSZ 64

typedef __attribute__((ext_vector_type(16))) __bf16 v16bf;
typedef __attribute__((ext_vector_type(8)))  float  v8f;

union Frag16 { v16bf v; int4 q[2]; };

__device__ __forceinline__ v8f wmma_bf16(v16bf a, v16bf b, v8f c) {
  // D = A(16x32 bf16) * B(32x16 bf16) + C(16x16 f32)
  return __builtin_amdgcn_wmma_f32_16x16x32_bf16(false, a, false, b, (short)0, c, false, false);
}

__device__ __forceinline__ unsigned short f2bf(float f) {
  unsigned int u = __float_as_uint(f);
  u += 0x7FFFu + ((u >> 16) & 1u);   // round-to-nearest-even
  return (unsigned short)(u >> 16);
}

// ---------------- async global->LDS (CDNA5 GLOBAL_LOAD_ASYNC_TO_LDS_B128) ----------------
#if defined(__AMDGCN__) && __has_builtin(__builtin_amdgcn_global_load_async_to_lds_b128)
#define HAS_ASYNC_LDS 1
typedef int v4i_gcc __attribute__((vector_size(16)));
typedef __attribute__((address_space(1))) v4i_gcc as1_v4i;
typedef __attribute__((address_space(3))) v4i_gcc as3_v4i;
#else
#define HAS_ASYNC_LDS 0
#endif

__device__ __forceinline__ void g2l_b128(const unsigned short* g, unsigned short* l) {
#if HAS_ASYNC_LDS
  __builtin_amdgcn_global_load_async_to_lds_b128((as1_v4i*)g, (as3_v4i*)l, 0, 0);
#else
  *reinterpret_cast<int4*>(l) = *reinterpret_cast<const int4*>(g);
#endif
}

template <int N>
__device__ __forceinline__ void wait_async() {
#if HAS_ASYNC_LDS
#if __has_builtin(__builtin_amdgcn_s_wait_asynccnt)
  __builtin_amdgcn_s_wait_asynccnt((short)N);
#else
  asm volatile("s_wait_asynccnt %0" :: "i"(N) : "memory");
#endif
#endif
}

// ---------------------------------------------------------------- convert
__global__ void cvt_bf16(const float4* __restrict__ in, ushort4* __restrict__ out, int n4) {
  int i = blockIdx.x * blockDim.x + threadIdx.x;
  if (i < n4) {
    float4 f = in[i];
    ushort4 r;
    r.x = f2bf(f.x); r.y = f2bf(f.y); r.z = f2bf(f.z); r.w = f2bf(f.w);
    out[i] = r;
  }
}

// ---------------------------------------------------------------- shared GEMM mainloop
// acc[c] = 16x16 f32 tile at (wave rows, ntile + c*16); K-tile = 64, double buffered.
__device__ __forceinline__ void load_x_tile(const unsigned short* __restrict__ Xb,
                                            int mtile, int kb, unsigned short* dstbase, int t) {
  int row = t >> 1;               // 0..127
  int ko  = (t & 1) * 32;         // elements
  const unsigned short* src = Xb + (size_t)(mtile + row) * DMODEL + kb + ko;
  unsigned short* dst = dstbase + row * 64 + ko;
  g2l_b128(src,      dst);
  g2l_b128(src + 8,  dst + 8);
  g2l_b128(src + 16, dst + 16);
  g2l_b128(src + 24, dst + 24);
}

__device__ __forceinline__ void load_w_tile(const unsigned short* __restrict__ W,
                                            int nloc, int kb, unsigned short* dstbase, int t) {
  int row = t >> 2;               // 0..63
  int ko  = (t & 3) * 16;
  const unsigned short* src = W + (size_t)(nloc + row) * DMODEL + kb + ko;
  unsigned short* dst = dstbase + row * 64 + ko;
  g2l_b128(src,     dst);
  g2l_b128(src + 8, dst + 8);
}

__device__ __forceinline__ void gemm_mainloop(
    const unsigned short* __restrict__ Xb, const unsigned short* __restrict__ W,
    int mtile, int nloc, int t,
    unsigned short (&xs)[2][128 * 64], unsigned short (&wt)[2][64 * 64],
    v8f (&acc)[4])
{
  const int w = t >> 5, l = t & 31, lm = l & 15, hf = l >> 4;
  const int arow = w * 16 + lm;
  const int NK = DMODEL / 64;     // 32 iterations

  // prologue: prefetch tile 0
  load_x_tile(Xb, mtile, 0, xs[0], t);
  load_w_tile(W,  nloc,  0, wt[0], t);

  for (int it = 0; it < NK; ++it) {
    const int buf = it & 1;
    wait_async<0>();              // own prefetch of tile `it` done
    __syncthreads();              // all waves done loading & done reading buf^1
    if (it + 1 < NK) {            // prefetch next tile (overlaps WMMAs below)
      load_x_tile(Xb, mtile, (it + 1) * 64, xs[buf ^ 1], t);
      load_w_tile(W,  nloc,  (it + 1) * 64, wt[buf ^ 1], t);
    }
    const unsigned short* xp = xs[buf];
    const unsigned short* wp = wt[buf];
    #pragma unroll
    for (int s = 0; s < 2; ++s) {
      Frag16 a, b[4];
      a.q[0] = *reinterpret_cast<const int4*>(xp + arow * 64 + s * 32 + hf * 8);
      a.q[1] = *reinterpret_cast<const int4*>(xp + arow * 64 + s * 32 + 16 + hf * 8);
      #pragma unroll
      for (int c = 0; c < 4; ++c) {
        b[c].q[0] = *reinterpret_cast<const int4*>(wp + (c * 16 + lm) * 64 + s * 32 + hf * 8);
        b[c].q[1] = *reinterpret_cast<const int4*>(wp + (c * 16 + lm) * 64 + s * 32 + 16 + hf * 8);
      }
      #pragma unroll
      for (int c = 0; c < 4; ++c) acc[c] = wmma_bf16(a.v, b[c].v, acc[c]);
    }
  }
}

// ---------------------------------------------------------------- QKV GEMM + RoPE
__global__ __launch_bounds__(256) void qkv_rope_gemm(
    const unsigned short* __restrict__ Xb,    // [4096][2048] bf16
    const unsigned short* __restrict__ Wqb,   // [2048][2048]
    const unsigned short* __restrict__ Wkb,   // [512][2048]
    const unsigned short* __restrict__ Wvb,   // [512][2048]
    const float* __restrict__ cosT,           // [2048][64]
    const float* __restrict__ sinT,
    unsigned short* __restrict__ Qo,          // [B][H][L][64]
    unsigned short* __restrict__ Ko,          // [B][G][L][64]
    unsigned short* __restrict__ Vo)          // [B][G][L][64]
{
  __shared__ alignas(16) unsigned short xs[2][128 * 64];
  __shared__ alignas(16) unsigned short wt[2][64 * 64];

  const int t  = threadIdx.x;
  const int w  = t >> 5, l = t & 31, lm = l & 15, hf = l >> 4;
  const int mtile = blockIdx.y * 128;
  const int ntile = blockIdx.x * 64;

  const unsigned short* W; int nloc, kind;
  if (ntile < DMODEL)            { W = Wqb; nloc = ntile;        kind = 0; }
  else if (ntile < DMODEL + 512) { W = Wkb; nloc = ntile - 2048; kind = 1; }
  else                           { W = Wvb; nloc = ntile - 2560; kind = 2; }

  v8f acc[4];
  #pragma unroll
  for (int c = 0; c < 4; ++c)
    #pragma unroll
    for (int g = 0; g < 8; ++g) acc[c][g] = 0.f;

  gemm_mainloop(Xb, W, mtile, nloc, t, xs, wt, acc);

  // epilogue: RoPE (q,k) + scale(q), scatter into head-major layouts
  const int gmBase = mtile + w * 16 + 8 * hf;
  #pragma unroll
  for (int g = 0; g < 8; ++g) {
    int gm   = gmBase + g;
    int bidx = gm >> 11;
    int pos  = gm & 2047;
    #pragma unroll
    for (int c = 0; c < 4; ++c) {
      int nc = ntile + c * 16 + lm;
      int hs = nc & 63;
      float v = acc[c][g];
      if (kind == 2) {
        int h = (nc - 2560) >> 6;
        Vo[(((size_t)bidx * NKV + h) * L_SEQ + pos) * HSZ + hs] = f2bf(v);
      } else {
        float cs = cosT[pos * HSZ + hs];
        float sn = sinT[pos * HSZ + hs];
        float partner = acc[c ^ 2][g];
        float rot = (hs < 32) ? -partner : partner;
        float r = v * cs + rot * sn;
        if (kind == 0) {
          r *= 0.125f;                       // 1/sqrt(HS) folded into Q
          int h = nc >> 6;
          Qo[(((size_t)bidx * NHEAD + h) * L_SEQ + pos) * HSZ + hs] = f2bf(r);
        } else {
          int h = (nc - 2048) >> 6;
          Ko[(((size_t)bidx * NKV + h) * L_SEQ + pos) * HSZ + hs] = f2bf(r);
        }
      }
    }
  }
}

// ---------------------------------------------------------------- flash attention
__global__ __launch_bounds__(256) void flash_attn(
    const unsigned short* __restrict__ Q,   // [B][H][L][64]
    const unsigned short* __restrict__ K,   // [B][G][L][64]
    const unsigned short* __restrict__ V,   // [B][G][L][64]
    unsigned short* __restrict__ O)         // [B][L][2048]
{
  __shared__ alignas(16) unsigned short Qs[128 * 64];
  __shared__ alignas(16) unsigned short Ks[64 * 64];
  __shared__ alignas(16) unsigned short Vt[64 * 64];      // transposed [hs][key]
  __shared__ alignas(16) unsigned short Ps[8][16 * 64];   // per-wave P staging

  const int t  = threadIdx.x;
  const int w  = t >> 5, l = t & 31, lm = l & 15, hf = l >> 4;
  const int qb = blockIdx.x * 128;
  const int bh = blockIdx.y;
  const int b  = bh >> 5, h = bh & 31;
  const int gh = h >> 2;                                  // GQA: 4 Q heads per KV head

  const unsigned short* Qbase = Q + (((size_t)b * NHEAD + h) * L_SEQ + qb) * HSZ;
  const unsigned short* Kbase = K + (((size_t)b * NKV + gh) * L_SEQ) * HSZ;
  const unsigned short* Vbase = V + (((size_t)b * NKV + gh) * L_SEQ) * HSZ;

  { // Q tile 128x64 (once), async path
    int row = t >> 1, ko = (t & 1) * 32;
    const unsigned short* src = Qbase + row * HSZ + ko;
    unsigned short* dst = Qs + row * 64 + ko;
    g2l_b128(src,      dst);
    g2l_b128(src + 8,  dst + 8);
    g2l_b128(src + 16, dst + 16);
    g2l_b128(src + 24, dst + 24);
  }
  wait_async<0>();
  __syncthreads();

  Frag16 aq[2];
  {
    const int arow = w * 16 + lm;
    #pragma unroll
    for (int s = 0; s < 2; ++s) {
      aq[s].q[0] = *reinterpret_cast<const int4*>(Qs + arow * 64 + s * 32 + hf * 8);
      aq[s].q[1] = *reinterpret_cast<const int4*>(Qs + arow * 64 + s * 32 + 16 + hf * 8);
    }
  }

  float m_run[8], l_run[8];
  #pragma unroll
  for (int g = 0; g < 8; ++g) { m_run[g] = -3.0e38f; l_run[g] = 0.f; }
  v8f o[4];
  #pragma unroll
  for (int c = 0; c < 4; ++c)
    #pragma unroll
    for (int g = 0; g < 8; ++g) o[c][g] = 0.f;

  const int wbrow = qb + w * 16;
  const int nkt   = (qb + 128) >> 6;

  for (int kt = 0; kt < nkt; ++kt) {
    const int ktb = kt << 6;
    __syncthreads();
    { // K tile natural [key][hs] — async direct-to-LDS
      int row = t >> 2, ko = (t & 3) * 16;
      const unsigned short* src = Kbase + (size_t)(ktb + row) * HSZ + ko;
      unsigned short* dst = Ks + row * 64 + ko;
      g2l_b128(src,     dst);
      g2l_b128(src + 8, dst + 8);
    }
    { // V tile transposed [hs][key] — needs element transpose, sync path
      int key = t >> 2, ho = (t & 3) * 16;
      union { int4 q[2]; unsigned short u[16]; } tmp;
      const int4* src = reinterpret_cast<const int4*>(Vbase + (size_t)(ktb + key) * HSZ + ho);
      tmp.q[0] = src[0]; tmp.q[1] = src[1];
      #pragma unroll
      for (int i = 0; i < 16; ++i) Vt[(ho + i) * 64 + key] = tmp.u[i];
    }
    wait_async<0>();
    __syncthreads();

    if (ktb <= wbrow + 15) {           // wave-uniform: tile has unmasked keys
      v8f s[4];
      #pragma unroll
      for (int c = 0; c < 4; ++c) {
        #pragma unroll
        for (int g = 0; g < 8; ++g) s[c][g] = 0.f;
        #pragma unroll
        for (int ks = 0; ks < 2; ++ks) {
          Frag16 bk;
          const int brow = c * 16 + lm;
          bk.q[0] = *reinterpret_cast<const int4*>(Ks + brow * 64 + ks * 32 + hf * 8);
          bk.q[1] = *reinterpret_cast<const int4*>(Ks + brow * 64 + ks * 32 + 16 + hf * 8);
          s[c] = wmma_bf16(aq[ks].v, bk.v, s[c]);
        }
      }
      if (ktb + 63 > wbrow) {          // causal mask on diagonal tiles
        #pragma unroll
        for (int c = 0; c < 4; ++c) {
          int col = ktb + c * 16 + lm;
          #pragma unroll
          for (int g = 0; g < 8; ++g)
            if (col > wbrow + g + 8 * hf) s[c][g] = -1.0e30f;
        }
      }
      // online softmax (row g+8*hf lives within one 16-lane half -> xor 1/2/4/8)
      float alpha[8];
      #pragma unroll
      for (int g = 0; g < 8; ++g) {
        float m = fmaxf(fmaxf(s[0][g], s[1][g]), fmaxf(s[2][g], s[3][g]));
        m = fmaxf(m, __shfl_xor(m, 1));
        m = fmaxf(m, __shfl_xor(m, 2));
        m = fmaxf(m, __shfl_xor(m, 4));
        m = fmaxf(m, __shfl_xor(m, 8));
        float mn = fmaxf(m_run[g], m);
        alpha[g] = __expf(m_run[g] - mn);
        m_run[g] = mn;
      }
      float psum[8];
      #pragma unroll
      for (int g = 0; g < 8; ++g) psum[g] = 0.f;
      #pragma unroll
      for (int c = 0; c < 4; ++c)
        #pragma unroll
        for (int g = 0; g < 8; ++g) {
          float p = __expf(s[c][g] - m_run[g]);
          psum[g] += p;
          Ps[w][(g + 8 * hf) * 64 + c * 16 + lm] = f2bf(p);
        }
      #pragma unroll
      for (int g = 0; g < 8; ++g) {
        float ps = psum[g];
        ps += __shfl_xor(ps, 1);
        ps += __shfl_xor(ps, 2);
        ps += __shfl_xor(ps, 4);
        ps += __shfl_xor(ps, 8);
        l_run[g] = l_run[g] * alpha[g] + ps;
        #pragma unroll
        for (int c = 0; c < 4; ++c) o[c][g] *= alpha[g];
      }
      // O += P @ V  (Vt layout => contiguous B reads; Ps wave-private, same-wave LDS in-order)
      Frag16 ap[2];
      #pragma unroll
      for (int s2 = 0; s2 < 2; ++s2) {
        ap[s2].q[0] = *reinterpret_cast<const int4*>(&Ps[w][lm * 64 + s2 * 32 + hf * 8]);
        ap[s2].q[1] = *reinterpret_cast<const int4*>(&Ps[w][lm * 64 + s2 * 32 + 16 + hf * 8]);
      }
      #pragma unroll
      for (int c = 0; c < 4; ++c) {
        #pragma unroll
        for (int s2 = 0; s2 < 2; ++s2) {
          Frag16 bv;
          const int brow = c * 16 + lm;
          bv.q[0] = *reinterpret_cast<const int4*>(Vt + brow * 64 + s2 * 32 + hf * 8);
          bv.q[1] = *reinterpret_cast<const int4*>(Vt + brow * 64 + s2 * 32 + 16 + hf * 8);
          o[c] = wmma_bf16(ap[s2].v, bv.v, o[c]);
        }
      }
    }
  }

  #pragma unroll
  for (int g = 0; g < 8; ++g) {
    int pos = qb + w * 16 + g + 8 * hf;
    float inv = 1.f / l_run[g];
    #pragma unroll
    for (int c = 0; c < 4; ++c) {
      int hs = c * 16 + lm;
      O[((size_t)b * L_SEQ + pos) * DMODEL + h * HSZ + hs] = f2bf(o[c][g] * inv);
    }
  }
}

// ---------------------------------------------------------------- output projection
__global__ __launch_bounds__(256) void out_proj(
    const unsigned short* __restrict__ Ob,    // [4096][2048] bf16
    const unsigned short* __restrict__ Wob,   // [2048][2048] bf16
    float* __restrict__ out)                  // [4096][2048] f32
{
  __shared__ alignas(16) unsigned short xs[2][128 * 64];
  __shared__ alignas(16) unsigned short wt[2][64 * 64];

  const int t  = threadIdx.x;
  const int w  = t >> 5, l = t & 31, lm = l & 15, hf = l >> 4;
  const int mtile = blockIdx.y * 128;
  const int ntile = blockIdx.x * 64;

  v8f acc[4];
  #pragma unroll
  for (int c = 0; c < 4; ++c)
    #pragma unroll
    for (int g = 0; g < 8; ++g) acc[c][g] = 0.f;

  gemm_mainloop(Ob, Wob, mtile, ntile, t, xs, wt, acc);

  const int gmBase = mtile + w * 16 + 8 * hf;
  #pragma unroll
  for (int g = 0; g < 8; ++g) {
    size_t gm = gmBase + g;
    #pragma unroll
    for (int c = 0; c < 4; ++c)
      out[gm * DMODEL + ntile + c * 16 + lm] = acc[c][g];
  }
}

// ---------------------------------------------------------------- launch
extern "C" void kernel_launch(void* const* d_in, const int* in_sizes, int n_in,
                              void* d_out, int out_size, void* d_ws, size_t ws_size,
                              hipStream_t stream) {
  const float* x    = (const float*)d_in[0];
  const float* cosT = (const float*)d_in[1];
  const float* sinT = (const float*)d_in[2];
  const float* Wq   = (const float*)d_in[3];
  const float* Wk   = (const float*)d_in[4];
  const float* Wv   = (const float*)d_in[5];
  const float* Wo   = (const float*)d_in[6];
  float* out = (float*)d_out;

  char* ws = (char*)d_ws;
  size_t off = 0;
  auto alloc = [&](size_t bytes) -> char* {
    char* p = ws + off;
    off += (bytes + 255) & ~(size_t)255;
    return p;
  };
  unsigned short* xb  = (unsigned short*)alloc((size_t)4096 * 2048 * 2);
  unsigned short* wqb = (unsigned short*)alloc((size_t)2048 * 2048 * 2);
  unsigned short* wkb = (unsigned short*)alloc((size_t)512 * 2048 * 2);
  unsigned short* wvb = (unsigned short*)alloc((size_t)512 * 2048 * 2);
  unsigned short* wob = (unsigned short*)alloc((size_t)2048 * 2048 * 2);
  unsigned short* Qw  = (unsigned short*)alloc((size_t)2 * NHEAD * L_SEQ * HSZ * 2);
  unsigned short* Kw  = (unsigned short*)alloc((size_t)2 * NKV * L_SEQ * HSZ * 2);
  unsigned short* Vw  = (unsigned short*)alloc((size_t)2 * NKV * L_SEQ * HSZ * 2);
  unsigned short* Ow  = (unsigned short*)alloc((size_t)4096 * 2048 * 2);

  auto cvt = [&](const float* in, unsigned short* o, int n) {
    int n4 = n / 4;
    cvt_bf16<<<(n4 + 255) / 256, 256, 0, stream>>>((const float4*)in, (ushort4*)o, n4);
  };
  cvt(x,  xb,  4096 * 2048);
  cvt(Wq, wqb, 2048 * 2048);
  cvt(Wk, wkb, 512 * 2048);
  cvt(Wv, wvb, 512 * 2048);
  cvt(Wo, wob, 2048 * 2048);

  qkv_rope_gemm<<<dim3(48, 32), 256, 0, stream>>>(xb, wqb, wkb, wvb, cosT, sinT, Qw, Kw, Vw);
  flash_attn<<<dim3(16, 64), 256, 0, stream>>>(Qw, Kw, Vw, Ow);
  out_proj<<<dim3(32, 32), 256, 0, stream>>>(Ow, wob, out);
}
